// Ensemble_55783035240903
// MI455X (gfx1250) — compile-verified, matrix operand order checked by reading, not run on hardware
//
#include <hip/hip_runtime.h>
#include <hip/hip_bf16.h>
#include <math.h>

#define T_TOK   4096
#define DIMD    1024
#define DFFD    4096
#define VOCABV  32000
#define NLLM    3
#define THRESH  0.98f
#define EPSF    1e-8f

typedef __attribute__((ext_vector_type(16))) __bf16 v16bf;
typedef __attribute__((ext_vector_type(8)))  float  v8f;
typedef int v4i __attribute__((vector_size(16)));          // matches builtin param type
typedef __attribute__((address_space(1))) v4i gv4i;        // global (device) AS
typedef __attribute__((address_space(3))) v4i lv4i;        // LDS AS

union Frag { v16bf v; uint4 q[2]; };

// ---- feature detection: gfx1250 async global->LDS copies (ASYNCcnt) ----
#ifdef __has_builtin
#if __has_builtin(__builtin_amdgcn_global_load_async_to_lds_b128)
#define USE_ASYNC_LDS 1
#endif
#if __has_builtin(__builtin_amdgcn_s_wait_asynccnt)
#define WAIT_ASYNC() __builtin_amdgcn_s_wait_asynccnt(0)
#endif
#endif
#ifndef USE_ASYNC_LDS
#define USE_ASYNC_LDS 0
#endif
#ifndef WAIT_ASYNC
#define WAIT_ASYNC() asm volatile("s_wait_asynccnt 0x0" ::: "memory")
#endif

__device__ __forceinline__ unsigned short f2bf(float f) {
  unsigned int u = __float_as_uint(f);
  u += 0x7fffu + ((u >> 16) & 1u);           // round-to-nearest-even
  return (unsigned short)(u >> 16);
}

// branch-free tanh-GELU: 1+tanh(y) = 2 - 2/(exp(2y)+1); exact limits at +-inf
__device__ __forceinline__ float gelu_tanh(float x) {
  float y = 0.7978845608028654f * (x + 0.044715f * x * x * x);
  float e = __expf(2.0f * y);
  return 0.5f * x * (2.0f - 2.0f / (e + 1.0f));
}

__device__ __forceinline__ float wave_sum(float v) {
#pragma unroll
  for (int m = 16; m > 0; m >>= 1) v += __shfl_xor(v, m, 32);
  return v;
}

// ---------------- embedding gather + active init ----------------
__global__ void k_gather(const int* __restrict__ x, const float* __restrict__ emb,
                         float* __restrict__ h, int* __restrict__ active) {
  int tid = blockIdx.x * blockDim.x + threadIdx.x;   // < T_TOK*DIMD
  int t = tid >> 10, d = tid & 1023;
  h[tid] = emb[(size_t)x[t] * DIMD + d];
  if (d == 0) active[t] = 1;
}

// ---------------- f32 -> bf16 (optionally transposed) ----------------
__global__ void k_transpose_cvt(const float* __restrict__ in, unsigned short* __restrict__ out,
                                int R, int C) {  // in: [R,C] f32, out: [C,R] bf16
  int idx = blockIdx.x * blockDim.x + threadIdx.x;   // < R*C
  int c = idx / R, r = idx - c * R;
  out[idx] = f2bf(in[(size_t)r * C + c]);
}

__global__ void k_cvt(const float* __restrict__ in, unsigned short* __restrict__ out) {
  size_t i = (size_t)blockIdx.x * blockDim.x + threadIdx.x;
  out[i] = f2bf(in[i]);
}

// ---------------- one wave per token layernorm ----------------
__global__ void k_layernorm(const float* __restrict__ h, const float* __restrict__ g,
                            const float* __restrict__ b, unsigned short* __restrict__ hn) {
  int wave = threadIdx.x >> 5, lane = threadIdx.x & 31;
  int t = blockIdx.x * 8 + wave;
  const float* row = h + (size_t)t * DIMD;
  float vals[32];
  float s = 0.f, ss = 0.f;
#pragma unroll
  for (int j = 0; j < 32; ++j) {
    float v = row[lane + 32 * j];
    vals[j] = v; s += v; ss += v * v;
  }
  s = wave_sum(s); ss = wave_sum(ss);
  float m   = s * (1.0f / DIMD);
  float var = ss * (1.0f / DIMD) - m * m;
  float inv = rsqrtf(var + 1e-5f);
  unsigned short* orow = hn + (size_t)t * DIMD;
#pragma unroll
  for (int j = 0; j < 32; ++j) {
    int d = lane + 32 * j;
    orow[d] = f2bf((vals[j] - m) * inv * g[d] + b[d]);
  }
}

// ---------------- cosine early-exit bookkeeping ----------------
__global__ void k_exit(float* __restrict__ h, const float* __restrict__ hout,
                       unsigned short* __restrict__ hexit, int* __restrict__ active, int last) {
  int wave = threadIdx.x >> 5, lane = threadIdx.x & 31;
  int t = blockIdx.x * 8 + wave;
  float* hr = h + (size_t)t * DIMD;
  const float* orow = hout + (size_t)t * DIMD;
  float d0 = 0.f, na = 0.f, nb = 0.f;
  float ov[32];
#pragma unroll
  for (int j = 0; j < 32; ++j) {
    float a = hr[lane + 32 * j], o = orow[lane + 32 * j];
    ov[j] = o; d0 += a * o; na += a * a; nb += o * o;
  }
  d0 = wave_sum(d0); na = wave_sum(na); nb = wave_sum(nb);
  float cosv = d0 / (sqrtf(na) * sqrtf(nb) + EPSF);
  int act  = active[t];
  int take = act && (last || cosv >= THRESH);
  if (take) {
    unsigned short* er = hexit + (size_t)t * DIMD;
#pragma unroll
    for (int j = 0; j < 32; ++j) er[lane + 32 * j] = f2bf(ov[j]);
    if (lane == 0) active[t] = 0;
  } else if (act) {
#pragma unroll
    for (int j = 0; j < 32; ++j) hr[lane + 32 * j] = ov[j];
  }
}

// ---------------- bf16 NT WMMA GEMM: C[M,N] = A[M,K] * B[N,K]^T ----------------
// Block: 256 thr (8 waves) -> 64x256 C tile; wave -> 32x64 (2x4 wmma accs).
// EPI: 0 = plain f32 store, 1 = bias+gelu -> bf16, 2 = bias+residual -> f32.
#define BM 64
#define BN 256
#define LDSROW 5                       // uint4 per LDS row: 64B data + 16B pad
#define LDSBUF ((BM + BN) * LDSROW)

template <int EPI>
__launch_bounds__(256)
__global__ void k_gemm_nt(const unsigned short* __restrict__ A,
                          const unsigned short* __restrict__ B,
                          void* __restrict__ outp,
                          const float* __restrict__ bias,
                          const float* __restrict__ res,
                          int K, int ldc) {
  __shared__ uint4 lds4[2 * LDSBUF];
  const int tid  = threadIdx.x;
  const int lane = tid & 31, idx = lane & 15, hlf = lane >> 4;
  const int wave = tid >> 5, wm = wave >> 2, wn = wave & 3;
  const int tileM = blockIdx.y * BM, tileN = blockIdx.x * BN;
  const int arow = tid >> 2, achk = tid & 3;   // 64 rows x 4 x 16B chunks

  v8f acc[2][4];
  v8f zero = {0.f, 0.f, 0.f, 0.f, 0.f, 0.f, 0.f, 0.f};
#pragma unroll
  for (int mi = 0; mi < 2; ++mi)
#pragma unroll
    for (int ni = 0; ni < 4; ++ni) acc[mi][ni] = zero;

  const int tiles = K >> 5;

  auto compute = [&](int p) {
    const uint4* base = &lds4[p * LDSBUF];
    Frag a[2], b[4];
#pragma unroll
    for (int mi = 0; mi < 2; ++mi) {
      int r = wm * 32 + mi * 16 + idx;
      a[mi].q[0] = base[r * LDSROW + hlf];             // K in [8*hlf, 8*hlf+8)
      a[mi].q[1] = base[r * LDSROW + 2 + hlf];         // K in [16+8*hlf, ...)
    }
#pragma unroll
    for (int ni = 0; ni < 4; ++ni) {
      int r = BM + wn * 64 + ni * 16 + idx;
      b[ni].q[0] = base[r * LDSROW + hlf];
      b[ni].q[1] = base[r * LDSROW + 2 + hlf];
    }
#pragma unroll
    for (int mi = 0; mi < 2; ++mi)
#pragma unroll
      for (int ni = 0; ni < 4; ++ni)
        acc[mi][ni] = __builtin_amdgcn_wmma_f32_16x16x32_bf16(
            false, a[mi].v, false, b[ni].v, (short)0, acc[mi][ni], false, false);
  };

#if USE_ASYNC_LDS
  // Zero-VGPR-staging pipeline: global_load_async_to_lds_b128 (ASYNCcnt),
  // overlapped with the WMMAs of the previous K-slab.
  auto issue = [&](int kt, int p) {
    uint4* da = &lds4[p * LDSBUF + arow * LDSROW + achk];
    const uint4* sa =
        reinterpret_cast<const uint4*>(A + (size_t)(tileM + arow) * K + (size_t)kt * 32) + achk;
    __builtin_amdgcn_global_load_async_to_lds_b128((gv4i*)sa, (lv4i*)da, 0, 0);
#pragma unroll
    for (int j = 0; j < 4; ++j) {
      int row = arow + j * 64;
      uint4* db = &lds4[p * LDSBUF + (BM + row) * LDSROW + achk];
      const uint4* sb =
          reinterpret_cast<const uint4*>(B + (size_t)(tileN + row) * K + (size_t)kt * 32) + achk;
      __builtin_amdgcn_global_load_async_to_lds_b128((gv4i*)sb, (lv4i*)db, 0, 0);
    }
  };

  issue(0, 0);
  WAIT_ASYNC();
  __syncthreads();
  for (int kt = 0; kt < tiles; ++kt) {
    const int p = kt & 1;
    if (kt + 1 < tiles) issue(kt + 1, p ^ 1);          // in flight behind WMMAs
    if (kt + 2 < tiles)                                // global_prefetch_b8
      __builtin_prefetch(B + (size_t)(tileN + arow) * K + (size_t)(kt + 2) * 32, 0, 1);
    compute(p);
    if (kt + 1 < tiles) WAIT_ASYNC();
    __syncthreads();
  }
#else
  // Fallback: VGPR staging + ds_store.
  uint4 ra; uint4 rb[4];
  auto gload = [&](int kt) {
    const uint4* pa = reinterpret_cast<const uint4*>(A + (size_t)(tileM + arow) * K + (size_t)kt * 32);
    ra = pa[achk];
#pragma unroll
    for (int j = 0; j < 4; ++j) {
      int row = arow + j * 64;
      const uint4* pb = reinterpret_cast<const uint4*>(B + (size_t)(tileN + row) * K + (size_t)kt * 32);
      rb[j] = pb[achk];
    }
  };
  auto stage = [&](int p) {
    uint4* base = &lds4[p * LDSBUF];
    base[arow * LDSROW + achk] = ra;
#pragma unroll
    for (int j = 0; j < 4; ++j) {
      int row = arow + j * 64;
      base[(BM + row) * LDSROW + achk] = rb[j];
    }
  };
  gload(0);
  stage(0);
  __syncthreads();
  for (int kt = 0; kt < tiles; ++kt) {
    const int p = kt & 1;
    if (kt + 1 < tiles) gload(kt + 1);
    if (kt + 2 < tiles)
      __builtin_prefetch(B + (size_t)(tileN + arow) * K + (size_t)(kt + 2) * 32, 0, 1);
    compute(p);
    if (kt + 1 < tiles) stage(p ^ 1);
    __syncthreads();
  }
#endif

  // epilogue: lane (idx,hlf) owns col n=idx, rows r+8*hlf of each 16x16 tile
#pragma unroll
  for (int mi = 0; mi < 2; ++mi) {
#pragma unroll
    for (int ni = 0; ni < 4; ++ni) {
      int n = tileN + wn * 64 + ni * 16 + idx;
      int mbase = tileM + wm * 32 + mi * 16 + hlf * 8;
      v8f c = acc[mi][ni];
#pragma unroll
      for (int r = 0; r < 8; ++r) {
        int m = mbase + r;
        float v = c[r];
        if (EPI == 1) {
          v = gelu_tanh(v + bias[n]);
          ((unsigned short*)outp)[(size_t)m * ldc + n] = f2bf(v);
        } else if (EPI == 2) {
          ((float*)outp)[(size_t)m * ldc + n] = v + bias[n] + res[(size_t)m * ldc + n];
        } else {
          ((float*)outp)[(size_t)m * ldc + n] = v;
        }
      }
    }
  }
}

// ---------------------------------------------------------------------------
extern "C" void kernel_launch(void* const* d_in, const int* in_sizes, int n_in,
                              void* d_out, int out_size, void* d_ws, size_t ws_size,
                              hipStream_t stream) {
  const int*   x    = (const int*)d_in[0];
  const float* emb  = (const float*)d_in[1];
  const float* ln_g = (const float*)d_in[2];
  const float* ln_b = (const float*)d_in[3];
  const float* W1   = (const float*)d_in[4];
  const float* b1   = (const float*)d_in[5];
  const float* W2   = (const float*)d_in[6];
  const float* b2   = (const float*)d_in[7];
  const float* Wout = (const float*)d_in[8];
  float* out = (float*)d_out;
  (void)in_sizes; (void)n_in; (void)out_size; (void)ws_size;

  char* ws = (char*)d_ws;
  size_t off = 0;
  auto wsalloc = [&](size_t bytes) {
    char* p = ws + off;
    off = (off + bytes + 255) & ~(size_t)255;
    return p;
  };
  float*          h     = (float*)wsalloc((size_t)T_TOK * DIMD * 4);
  float*          hout  = (float*)wsalloc((size_t)T_TOK * DIMD * 4);
  unsigned short* hn    = (unsigned short*)wsalloc((size_t)T_TOK * DIMD * 2);
  unsigned short* actb  = (unsigned short*)wsalloc((size_t)T_TOK * DFFD * 2);
  unsigned short* hexit = (unsigned short*)wsalloc((size_t)T_TOK * DIMD * 2);
  unsigned short* w1t   = (unsigned short*)wsalloc((size_t)DFFD * DIMD * 2);   // per-layer reuse
  unsigned short* w2t   = (unsigned short*)wsalloc((size_t)DIMD * DFFD * 2);   // per-layer reuse
  unsigned short* woutb = (unsigned short*)wsalloc((size_t)VOCABV * DIMD * 2);
  int*            active = (int*)wsalloc((size_t)T_TOK * 4);

  k_gather<<<(T_TOK * DIMD) / 256, 256, 0, stream>>>(x, emb, h, active);
  k_cvt<<<(VOCABV * DIMD) / 256, 256, 0, stream>>>(Wout, woutb);

  for (int i = 0; i < NLLM; ++i) {
    // W1[i]: [D,DFF] -> w1t [DFF,D] (NT);  W2[i]: [DFF,D] -> w2t [D,DFF] (NT)
    k_transpose_cvt<<<(DIMD * DFFD) / 256, 256, 0, stream>>>(W1 + (size_t)i * DIMD * DFFD, w1t, DIMD, DFFD);
    k_transpose_cvt<<<(DIMD * DFFD) / 256, 256, 0, stream>>>(W2 + (size_t)i * DFFD * DIMD, w2t, DFFD, DIMD);
    k_layernorm<<<T_TOK / 8, 256, 0, stream>>>(h, ln_g + i * DIMD, ln_b + i * DIMD, hn);
    // act = gelu(hn @ W1 + b1)  [T,DFF] bf16
    k_gemm_nt<1><<<dim3(DFFD / BN, T_TOK / BM), 256, 0, stream>>>(hn, w1t, actb, b1 + i * DFFD, nullptr, DIMD, DFFD);
    // h_out = h + act @ W2 + b2  [T,D] f32
    k_gemm_nt<2><<<dim3(DIMD / BN, T_TOK / BM), 256, 0, stream>>>(actb, w2t, hout, b2 + i * DIMD, h, DFFD, DIMD);
    k_exit<<<T_TOK / 8, 256, 0, stream>>>(h, hout, hexit, active, (i == NLLM - 1) ? 1 : 0);
  }
  // single shared head GEMM over each token's exit-time hidden state
  k_gemm_nt<0><<<dim3(VOCABV / BN, T_TOK / BM), 256, 0, stream>>>(hexit, woutb, out, nullptr, nullptr, DIMD, VOCABV);
}
